// Decoder_1511828488870
// MI455X (gfx1250) — matrix operand order repaired and checked by reading, not compiled, not run on hardware
//
#include <hip/hip_runtime.h>

typedef __attribute__((ext_vector_type(16))) _Float16 v16h;
typedef __attribute__((ext_vector_type(8)))  _Float16 v8h;
typedef __attribute__((ext_vector_type(8)))  float    v8f;

#define N_NODES 768
#define EMBED_DIM 128
#define N_EDGES 50000

// workspace layout (bytes)
#define WS_ACC    0          // 1 f32 accumulator
#define WS_PQ     256        // 768*256 f32 = 786432  (P = [:, :128], Q = [:, 128:])
#define WS_CMASK  786688     // 768*768 f32 = 2359296 (mask/od')
#define WS_E16    3145984    // 768*128 f16
#define WS_WCATT  3342592    // 256*128 f16 (WcatT[c][k])
#define WS_W2BT   3408128    // 64*128  f16 (W2bT[n][k])

// ---------------------------------------------------------------------------
// Phase 0: conversions + accumulator zero
// ---------------------------------------------------------------------------
__global__ void k_convert(const float* __restrict__ emb,
                          const float* __restrict__ W2a,
                          const float* __restrict__ W2b,
                          _Float16* __restrict__ E16,
                          _Float16* __restrict__ WcatT,
                          _Float16* __restrict__ W2bT,
                          float* __restrict__ acc) {
  int gid = blockIdx.x * blockDim.x + threadIdx.x;
  if (gid == 0) acc[0] = 0.0f;
  if (gid < N_NODES * EMBED_DIM) {
    E16[gid] = (_Float16)emb[gid];
  } else if (gid < N_NODES * EMBED_DIM + 256 * 128) {
    int t = gid - N_NODES * EMBED_DIM;
    int c = t >> 7, k = t & 127;
    float v = (c < 128) ? W2a[k * 128 + c] : W2a[(128 + k) * 128 + (c - 128)];
    WcatT[t] = (_Float16)v;
  } else if (gid < N_NODES * EMBED_DIM + 256 * 128 + 64 * 128) {
    int t = gid - N_NODES * EMBED_DIM - 256 * 128;
    int n = t >> 7, k = t & 127;
    W2bT[t] = (_Float16)W2b[k * 64 + n];
  }
}

__global__ void k_cmask(const float* __restrict__ od, float* __restrict__ C) {
  int gid = blockIdx.x * blockDim.x + threadIdx.x;
  if (gid < N_NODES * N_NODES) {
    float v = od[gid];
    v = (v == 0.0f) ? 1.0f : v;
    C[gid] = 1.0f / v;
  }
}

__global__ void k_scatter(const int* __restrict__ o_nodes,
                          const int* __restrict__ d_nodes,
                          float* __restrict__ C) {
  int gid = blockIdx.x * blockDim.x + threadIdx.x;
  if (gid < N_EDGES) C[o_nodes[gid] * N_NODES + d_nodes[gid]] = 0.0f;
}

// ---------------------------------------------------------------------------
// Phase 1: PQ = E16 [768x128] @ Wcat [128x256]  (f16 WMMA, f32 out)
// block = 128 threads (4 waves), each wave owns one 16x16 output tile
// ---------------------------------------------------------------------------
__global__ void k_gemm_pq(const _Float16* __restrict__ E16,
                          const _Float16* __restrict__ WcatT,
                          float* __restrict__ PQ) {
  int tid = threadIdx.x;
  int wave = tid >> 5, lane = tid & 31;
  int l16 = lane & 15, h16 = lane >> 4;         // half-wave select
  int wt = blockIdx.x * 4 + wave;               // 0..767 wave-tiles
  int mtile = wt >> 4;                          // 0..47
  int ntile = wt & 15;                          // 0..15

  v8f c = {};
  const _Float16* arow = E16 + (mtile * 16 + l16) * 128;
  const _Float16* bcol = WcatT + (ntile * 16 + l16) * 128;
#pragma unroll
  for (int s = 0; s < 4; ++s) {
    int k0 = s * 32 + h16 * 16;                 // A: 16 contiguous K per lane
    v8h lo = *(const v8h*)(arow + k0);
    v8h hi = *(const v8h*)(arow + k0 + 8);
    v16h a = __builtin_shufflevector(lo, hi, 0,1,2,3,4,5,6,7,8,9,10,11,12,13,14,15);
    int kb = s * 32 + h16 * 16;                 // B: 16 contiguous K per lane
    v16h b = *(const v16h*)(bcol + kb);
    c = __builtin_amdgcn_wmma_f32_16x16x32_f16(false, a, false, b, (short)0, c,
                                               false, false);
  }
  int n = ntile * 16 + l16;
#pragma unroll
  for (int v = 0; v < 8; ++v) {
    int m = mtile * 16 + v + h16 * 8;
    PQ[m * 256 + n] = c[v];
  }
}

// ---------------------------------------------------------------------------
// Shared MLP tail (16-row variant, 128 threads / 4 waves, one n-tile each):
// lds_h1[16][128] f16 -> lds_part[4][16] per-row partials of
// sum_n lrelu(h1 @ W2b + b2b)[n] * W2c[n]
// ---------------------------------------------------------------------------
__device__ __forceinline__ void mlp_tail(const _Float16* __restrict__ lds_h1,
                                         const _Float16* __restrict__ W2bT,
                                         const float* __restrict__ b2b,
                                         const float* __restrict__ W2c,
                                         float* __restrict__ lds_part) {
  int tid = threadIdx.x;
  int wave = tid >> 5, lane = tid & 31;
  int l16 = lane & 15, h16 = lane >> 4;

  v8f c = {};
  const _Float16* arow = lds_h1 + l16 * 128;
  const _Float16* bcol = W2bT + (wave * 16 + l16) * 128;
#pragma unroll
  for (int s = 0; s < 4; ++s) {
    int k0 = s * 32 + h16 * 16;
    v8h lo = *(const v8h*)(arow + k0);
    v8h hi = *(const v8h*)(arow + k0 + 8);
    v16h a = __builtin_shufflevector(lo, hi, 0,1,2,3,4,5,6,7,8,9,10,11,12,13,14,15);
    int kb = s * 32 + h16 * 16;
    v16h b = *(const v16h*)(bcol + kb);
    c = __builtin_amdgcn_wmma_f32_16x16x32_f16(false, a, false, b, (short)0, c,
                                               false, false);
  }
  int n = wave * 16 + l16;
  float bb = b2b[n];
  float wc = W2c[n];
  float part[8];
#pragma unroll
  for (int v = 0; v < 8; ++v) {
    float h = c[v] + bb;
    h = (h > 0.0f) ? h : 0.01f * h;             // LeakyReLU
    float t = h * wc;
    t += __shfl_xor(t, 1, 16);                  // reduce over 16 N-lanes
    t += __shfl_xor(t, 2, 16);
    t += __shfl_xor(t, 4, 16);
    t += __shfl_xor(t, 8, 16);
    part[v] = t;
  }
  if (l16 == 0) {
    int rowbase = h16 * 8;                      // C layout: M = v + 8*h16
#pragma unroll
    for (int v = 0; v < 8; ++v) lds_part[wave * 16 + rowbase + v] = part[v];
  }
}

// ---------------------------------------------------------------------------
// Negatives: block = 16 i x 16 j pair tile, 256 threads (8 waves).
// Wave w owns n-tile (w&3) with W2b B-fragments resident in registers,
// and M-tiles [(w>>2)*8, (w>>2)*8+8).
// ---------------------------------------------------------------------------
__global__ void __launch_bounds__(256) k_neg(
    const float* __restrict__ PQ,
    const float* __restrict__ b2a,
    const _Float16* __restrict__ W2bT,
    const float* __restrict__ b2b,
    const float* __restrict__ W2c,
    const float* __restrict__ b2c,
    const float* __restrict__ Cmask,
    float* __restrict__ acc) {
  __shared__ __align__(16) _Float16 lds_h1[256 * 128];  // 64 KB: h1 for 256 pairs
  __shared__ float lds_part[4][256];                    // per-ntile row partials

  int i0 = (blockIdx.x / 48) * 16;
  int j0 = (blockIdx.x % 48) * 16;
  int tid = threadIdx.x;

  // ---- stage 1: h1[p][k] = lrelu(P[i][k] + Q[j][k] + b2a[k]), p = pi*16+pj
  {
    int k = tid & 127;
    int ph = tid >> 7;                          // pi range: [ph*8, ph*8+8)
    float ba = b2a[k];
    float qv[16];
#pragma unroll
    for (int pj = 0; pj < 16; ++pj)
      qv[pj] = PQ[(j0 + pj) * 256 + 128 + k];
#pragma unroll
    for (int pi8 = 0; pi8 < 8; ++pi8) {
      int pi = ph * 8 + pi8;
      float pik = PQ[(i0 + pi) * 256 + k] + ba;
#pragma unroll
      for (int pj = 0; pj < 16; ++pj) {
        float pre = pik + qv[pj];
        lds_h1[(pi * 16 + pj) * 128 + k] =
            (_Float16)((pre > 0.0f) ? pre : 0.01f * pre);
      }
    }
  }
  __syncthreads();

  // ---- stage 2: layer 2+3 via WMMA, B-fragments register-resident
  {
    int wave = tid >> 5, lane = tid & 31;
    int l16 = lane & 15, h16 = lane >> 4;
    int ntile = wave & 3;
    int mbase = (wave >> 2) * 8;

    const _Float16* bcol = W2bT + (ntile * 16 + l16) * 128;
    v16h bfrag[4];
#pragma unroll
    for (int s = 0; s < 4; ++s)
      bfrag[s] = *(const v16h*)(bcol + s * 32 + h16 * 16);

    int n = ntile * 16 + l16;
    float bb = b2b[n];
    float wc = W2c[n];

#pragma unroll
    for (int mt = 0; mt < 8; ++mt) {
      int m = mbase + mt;
      const _Float16* arow = lds_h1 + (m * 16 + l16) * 128;
      v8f c = {};
#pragma unroll
      for (int s = 0; s < 4; ++s) {
        int k0 = s * 32 + h16 * 16;
        v8h lo = *(const v8h*)(arow + k0);
        v8h hi = *(const v8h*)(arow + k0 + 8);
        v16h a = __builtin_shufflevector(lo, hi,
                                         0,1,2,3,4,5,6,7,8,9,10,11,12,13,14,15);
        c = __builtin_amdgcn_wmma_f32_16x16x32_f16(false, a, false, bfrag[s],
                                                   (short)0, c, false, false);
      }
      float part[8];
#pragma unroll
      for (int v = 0; v < 8; ++v) {
        float h = c[v] + bb;
        h = (h > 0.0f) ? h : 0.01f * h;
        float t = h * wc;
        t += __shfl_xor(t, 1, 16);
        t += __shfl_xor(t, 2, 16);
        t += __shfl_xor(t, 4, 16);
        t += __shfl_xor(t, 8, 16);
        part[v] = t;
      }
      if (l16 == 0) {
#pragma unroll
        for (int v = 0; v < 8; ++v)
          lds_part[ntile][m * 16 + h16 * 8 + v] = part[v];
      }
    }
  }
  __syncthreads();

  // ---- final: out^2 * mask/od', wave reduce, one atomic per wave
  {
    float out = lds_part[0][tid] + lds_part[1][tid] + lds_part[2][tid] +
                lds_part[3][tid] + b2c[0];
    int pi = tid >> 4, pj = tid & 15;
    float l = out * out * Cmask[(i0 + pi) * N_NODES + (j0 + pj)];
    l += __shfl_xor(l, 1, 32);
    l += __shfl_xor(l, 2, 32);
    l += __shfl_xor(l, 4, 32);
    l += __shfl_xor(l, 8, 32);
    l += __shfl_xor(l, 16, 32);
    if ((tid & 31) == 0) atomicAdd(acc, l);
  }
}

// ---------------------------------------------------------------------------
// Positives: block handles 16 edges (50000 = 3125*16), 128 threads
// ---------------------------------------------------------------------------
__global__ void __launch_bounds__(128) k_pos(
    const float* __restrict__ PQ,
    const int* __restrict__ o_nodes,
    const int* __restrict__ d_nodes,
    const float* __restrict__ time_diffs,
    const float* __restrict__ edge_ind,
    const float* __restrict__ od,
    const float* __restrict__ w1,
    const float* __restrict__ b1,
    const float* __restrict__ b2a,
    const _Float16* __restrict__ W2bT,
    const float* __restrict__ b2b,
    const float* __restrict__ W2c,
    const float* __restrict__ b2c,
    const int* __restrict__ output_len,
    float* __restrict__ acc) {
  __shared__ __align__(16) _Float16 lds_h1[16 * 128];
  __shared__ float lds_part[64];

  int base = blockIdx.x * 16;
  int k = threadIdx.x;
  float ba = b2a[k];
  float inv_olen = 1.0f / (float)output_len[0];
  float w10 = w1[0], b10 = b1[0];

#pragma unroll
  for (int p = 0; p < 16; ++p) {
    int e = base + p;
    int o = o_nodes[e], d = d_nodes[e];
    float scale = time_diffs[e] * inv_olen * w10 + b10;
    float pre = scale * (PQ[o * 256 + k] + PQ[d * 256 + 128 + k]) + ba;
    lds_h1[p * 128 + k] = (_Float16)((pre > 0.0f) ? pre : 0.01f * pre);
  }
  __syncthreads();

  mlp_tail(lds_h1, W2bT, b2b, W2c, lds_part);
  __syncthreads();

  if (threadIdx.x < 16) {
    int p = threadIdx.x;
    int e = base + p;
    float out = lds_part[p] + lds_part[16 + p] + lds_part[32 + p] +
                lds_part[48 + p] + b2c[0];
    int o = o_nodes[e], d = d_nodes[e];
    float odv = od[o * N_NODES + d];
    odv = (odv == 0.0f) ? 1.0f : odv;
    float diff = out - edge_ind[e];
    float l = diff * diff / odv;
    l += __shfl_xor(l, 1, 16);
    l += __shfl_xor(l, 2, 16);
    l += __shfl_xor(l, 4, 16);
    l += __shfl_xor(l, 8, 16);
    if (p == 0) atomicAdd(acc, l);
  }
}

__global__ void k_final(const float* __restrict__ acc, float* __restrict__ out) {
  out[0] = acc[0] * (1.0f / ((float)N_NODES * (float)N_NODES));
}

// ---------------------------------------------------------------------------
extern "C" void kernel_launch(void* const* d_in, const int* in_sizes, int n_in,
                              void* d_out, int out_size, void* d_ws, size_t ws_size,
                              hipStream_t stream) {
  const float* emb       = (const float*)d_in[0];
  const int*   o_nodes   = (const int*)d_in[1];
  const int*   d_nodes   = (const int*)d_in[2];
  const float* time_diff = (const float*)d_in[3];
  const float* edge_ind  = (const float*)d_in[4];
  const float* od        = (const float*)d_in[5];
  const float* w1        = (const float*)d_in[6];
  const float* b1        = (const float*)d_in[7];
  const float* W2a       = (const float*)d_in[8];
  const float* b2a       = (const float*)d_in[9];
  const float* W2b       = (const float*)d_in[10];
  const float* b2b       = (const float*)d_in[11];
  const float* W2c       = (const float*)d_in[12];
  const float* b2c       = (const float*)d_in[13];
  const int*   olen      = (const int*)d_in[14];

  char* ws = (char*)d_ws;
  float*    acc   = (float*)(ws + WS_ACC);
  float*    PQ    = (float*)(ws + WS_PQ);
  float*    Cm    = (float*)(ws + WS_CMASK);
  _Float16* E16   = (_Float16*)(ws + WS_E16);
  _Float16* WcatT = (_Float16*)(ws + WS_WCATT);
  _Float16* W2bT  = (_Float16*)(ws + WS_W2BT);

  k_convert<<<544, 256, 0, stream>>>(emb, W2a, W2b, E16, WcatT, W2bT, acc);
  k_cmask<<<(N_NODES * N_NODES + 255) / 256, 256, 0, stream>>>(od, Cm);
  k_scatter<<<(N_EDGES + 255) / 256, 256, 0, stream>>>(o_nodes, d_nodes, Cm);
  k_gemm_pq<<<192, 128, 0, stream>>>(E16, WcatT, PQ);
  k_neg<<<48 * 48, 256, 0, stream>>>(PQ, b2a, W2bT, b2b, W2c, b2c, Cm, acc);
  k_pos<<<N_EDGES / 16, 128, 0, stream>>>(PQ, o_nodes, d_nodes, time_diff,
                                          edge_ind, od, w1, b1, b2a, W2bT, b2b,
                                          W2c, b2c, olen, acc);
  k_final<<<1, 1, 0, stream>>>(acc, (float*)d_out);
}
